// WindowAttention_46720654246523
// MI455X (gfx1250) — matrix operand order repaired and checked by reading, not compiled, not run on hardware
//
#include <hip/hip_runtime.h>
#include <hip/hip_bf16.h>

typedef __attribute__((ext_vector_type(16))) _Float16 v16h;
typedef __attribute__((ext_vector_type(8)))  _Float16 v8h;
typedef __attribute__((ext_vector_type(8)))  float    v8f;

#define D      256
#define NHEAD  8
#define DH     32
#define NWAVES 8

#define NW_A 2000
#define MT_A 36
#define NV_A 50000
#define NW_B 400
#define MT_B 144
#define NV_B 40000

// ---------------------------------------------------------------------------
// Fragment loaders (wave32, per CDNA5 ISA §7.12.2 layouts)
// A (16x32 f16): lane<16 holds row m0+lane, K chunks {k0..k0+7, k0+16..k0+23};
//                lane>=16 holds row m0+lane-16, K chunks {k0+8..15, k0+24..31}.
__device__ inline v16h load_a_frag(const _Float16* base, int stride, int m0, int k0) {
  const int l = threadIdx.x & 31;
  const int hh = l >> 4;
  const _Float16* p = base + (m0 + (l & 15)) * stride + k0 + hh * 8;
  v8h lo = *(const v8h*)(p);
  v8h hi = *(const v8h*)(p + 16);
  return __builtin_shufflevector(lo, hi, 0,1,2,3,4,5,6,7,8,9,10,11,12,13,14,15);
}
// B (32x16 f16, K x N): lane<16 holds col n0+lane, K chunks {k0..k0+7, k0+16..k0+23};
//                       lane>=16 holds col n0+lane-16, K chunks {k0+8..15, k0+24..31}.
// base is "N-major": row n holds the K values for column n (stride halves).
__device__ inline v16h load_b_frag(const _Float16* base, int stride, int n0, int k0) {
  const int l = threadIdx.x & 31;
  const int hh = l >> 4;
  const _Float16* p = base + (n0 + (l & 15)) * stride + k0 + hh * 8;
  v8h lo = *(const v8h*)(p);
  v8h hi = *(const v8h*)(p + 16);
  return __builtin_shufflevector(lo, hi, 0,1,2,3,4,5,6,7,8,9,10,11,12,13,14,15);
}

// ---------------------------------------------------------------------------
// Prologue kernels
__global__ void wa_init_s2r(int* s2r, int n) {
  int i = blockIdx.x * 256 + threadIdx.x;
  if (i < n) s2r[i] = -1;
}
__global__ void wa_scatter(int* s2rA, int* s2rB,
                           const int* idx_a, const int* idx_b,
                           const int* slot_a, const int* slot_b) {
  int i = blockIdx.x * 256 + threadIdx.x;
  if (i < NV_A) {
    s2rA[slot_a[i]] = idx_a[i];
  } else if (i < NV_A + NV_B) {
    int j = i - NV_A;
    s2rB[slot_b[j]] = idx_b[j];
  }
}
__global__ void wa_cvt_weights(const float* W_in, const float* W_out,
                               _Float16* Win16, _Float16* Wout16) {
  int i = blockIdx.x * 256 + threadIdx.x;
  if (i < 3 * D * D) Win16[i] = (_Float16)W_in[i];
  if (i < D * D)     Wout16[i] = (_Float16)W_out[i];
}

// ---------------------------------------------------------------------------
// Fused per-window MHA.  MT = real tokens/window, MTP = MT padded to 16,
// MTPS = MTP padded to 32 (V^T row stride, so odd-tile PV reads stay in range).
template <int MT, int MTP, int MTPS>
__global__ __launch_bounds__(256) void wa_attn(
    const float* __restrict__ feat, const float* __restrict__ pos,
    const float* __restrict__ b_in, const float* __restrict__ b_out,
    const int* __restrict__ s2r,
    const _Float16* __restrict__ Win16, const _Float16* __restrict__ Wout16,
    float* __restrict__ out) {
  constexpr int NT = D / 16;
  constexpr int MTILES = MTP / 16;

  const int tid = threadIdx.x;
  const int wv = tid >> 5;
  const int lane = tid & 31;
  const int l15 = lane & 15;
  const int half8 = (lane >= 16) ? 8 : 0;
  const int win = blockIdx.x;

  extern __shared__ _Float16 smem[];
  _Float16* bufA = smem;                 // x  -> later Q   (MTP x D)
  _Float16* bufB = bufA + MTP * D;       // qk -> later O   (MTP x D)
  _Float16* bufK = bufB + MTP * D;       // K               (MTP x D)
  _Float16* bufV = bufK + MTP * D;       // V^T             (D x MTPS)
  _Float16* pbuf = bufV + D * MTPS;      // per-wave P stage (8 x 16 x 32)
  float* maskb = (float*)(pbuf + NWAVES * 512);  // MTP mask biases

  // ---- Phase 0: gather x, build qk = x + pos (f16), masks, zero V^T pad ----
  for (int m = wv; m < MTP; m += NWAVES) {
    const int c0 = lane * 8;
    float xr[8], qr[8];
    int tok = -1;
    if (m < MT) tok = s2r[win * MT + m];
    if (tok >= 0) {
      const float4* fp = (const float4*)(feat + (long)tok * D + c0);
      float4 a0 = fp[0], a1 = fp[1];
      xr[0] = a0.x; xr[1] = a0.y; xr[2] = a0.z; xr[3] = a0.w;
      xr[4] = a1.x; xr[5] = a1.y; xr[6] = a1.z; xr[7] = a1.w;
    } else {
      #pragma unroll
      for (int i = 0; i < 8; ++i) xr[i] = 0.0f;
    }
    if (m < MT) {
      const float4* pp = (const float4*)(pos + ((long)win * MT + m) * D + c0);
      float4 p0 = pp[0], p1 = pp[1];
      qr[0] = xr[0] + p0.x; qr[1] = xr[1] + p0.y; qr[2] = xr[2] + p0.z; qr[3] = xr[3] + p0.w;
      qr[4] = xr[4] + p1.x; qr[5] = xr[5] + p1.y; qr[6] = xr[6] + p1.z; qr[7] = xr[7] + p1.w;
    } else {
      #pragma unroll
      for (int i = 0; i < 8; ++i) qr[i] = 0.0f;
    }
    v8h xh, qh;
    #pragma unroll
    for (int i = 0; i < 8; ++i) { xh[i] = (_Float16)xr[i]; qh[i] = (_Float16)qr[i]; }
    *(v8h*)(bufA + m * D + c0) = xh;
    *(v8h*)(bufB + m * D + c0) = qh;
  }
  if (tid < MTP) {
    bool occ = false;
    if (tid < MT) occ = (s2r[win * MT + tid] >= 0);
    maskb[tid] = occ ? 0.0f : -1.0e9f;
  }
  for (int i = tid; i < D * (MTPS - MTP); i += 256) {
    int r = i / (MTPS - MTP), c = MTP + i % (MTPS - MTP);
    bufV[r * MTPS + c] = (_Float16)0.0f;
  }
  __syncthreads();

  // ---- Phase 1: V^T = (x @ Wv^T)^T, f16 in LDS ----
  for (int t = wv; t < MTILES * NT; t += NWAVES) {
    const int mi = t / NT, ni = t % NT;
    const float bias = b_in[2 * D + ni * 16 + l15];
    v8f acc;
    #pragma unroll
    for (int r = 0; r < 8; ++r) acc[r] = bias;
    #pragma unroll
    for (int k = 0; k < D; k += 32) {
      v16h a = load_a_frag(bufA, D, mi * 16, k);
      v16h b = load_b_frag(Win16 + 2 * D * D, D, ni * 16, k);
      acc = __builtin_amdgcn_wmma_f32_16x16x32_f16(false, a, false, b, (short)0, acc, false, false);
    }
    #pragma unroll
    for (int r = 0; r < 8; ++r)
      bufV[(ni * 16 + l15) * MTPS + mi * 16 + half8 + r] = (_Float16)acc[r];
  }
  __syncthreads();  // bufA(x) consumed; Q may overwrite it

  // ---- Phase 2+3: Q = qk@Wq^T + bq (into bufA), K = qk@Wk^T + bk (bufK) ----
  for (int t = wv; t < MTILES * NT; t += NWAVES) {
    const int mi = t / NT, ni = t % NT;
    const float bq = b_in[ni * 16 + l15];
    v8f acc;
    #pragma unroll
    for (int r = 0; r < 8; ++r) acc[r] = bq;
    #pragma unroll
    for (int k = 0; k < D; k += 32) {
      v16h a = load_a_frag(bufB, D, mi * 16, k);
      v16h b = load_b_frag(Win16, D, ni * 16, k);
      acc = __builtin_amdgcn_wmma_f32_16x16x32_f16(false, a, false, b, (short)0, acc, false, false);
    }
    #pragma unroll
    for (int r = 0; r < 8; ++r)
      bufA[(mi * 16 + half8 + r) * D + ni * 16 + l15] = (_Float16)acc[r];
  }
  for (int t = wv; t < MTILES * NT; t += NWAVES) {
    const int mi = t / NT, ni = t % NT;
    const float bk = b_in[D + ni * 16 + l15];
    v8f acc;
    #pragma unroll
    for (int r = 0; r < 8; ++r) acc[r] = bk;
    #pragma unroll
    for (int k = 0; k < D; k += 32) {
      v16h a = load_a_frag(bufB, D, mi * 16, k);
      v16h b = load_b_frag(Win16 + D * D, D, ni * 16, k);
      acc = __builtin_amdgcn_wmma_f32_16x16x32_f16(false, a, false, b, (short)0, acc, false, false);
    }
    #pragma unroll
    for (int r = 0; r < 8; ++r)
      bufK[(mi * 16 + half8 + r) * D + ni * 16 + l15] = (_Float16)acc[r];
  }
  __syncthreads();  // qk(bufB) consumed; O may overwrite it

  // ---- Phase 4: per-head attention, wave w == head w ----
  {
    const int h = wv;
    _Float16* pb = pbuf + wv * 512;  // private 16x32 f16 stage
    const float rscale = 0.17677669529663687f;  // 1/sqrt(32)
    for (int mi = 0; mi < MTILES; ++mi) {
      v16h qf = load_a_frag(bufA, D, mi * 16, h * DH);
      v8f S[MTILES];
      #pragma unroll
      for (int nj = 0; nj < MTILES; ++nj) {
        v16h kf = load_b_frag(bufK, D, nj * 16, h * DH);
        v8f z;
        #pragma unroll
        for (int r = 0; r < 8; ++r) z[r] = 0.0f;
        z = __builtin_amdgcn_wmma_f32_16x16x32_f16(false, qf, false, kf, (short)0, z, false, false);
        const float mb = maskb[nj * 16 + l15];
        #pragma unroll
        for (int r = 0; r < 8; ++r) S[nj][r] = z[r] * rscale + mb;
      }
      // softmax across the row strip; rows split lanes0-15 / lanes16-31
      float rsum[8];
      #pragma unroll
      for (int r = 0; r < 8; ++r) {
        float mx = -3.0e38f;
        #pragma unroll
        for (int nj = 0; nj < MTILES; ++nj) mx = fmaxf(mx, S[nj][r]);
        #pragma unroll
        for (int off = 8; off >= 1; off >>= 1) mx = fmaxf(mx, __shfl_xor(mx, off, 16));
        float sm = 0.0f;
        #pragma unroll
        for (int nj = 0; nj < MTILES; ++nj) {
          float e = __expf(S[nj][r] - mx);
          S[nj][r] = e;
          sm += e;
        }
        #pragma unroll
        for (int off = 8; off >= 1; off >>= 1) sm += __shfl_xor(sm, off, 16);
        rsum[r] = sm;
      }
      // PV: restage P tiles (C->A fragment transpose through LDS), 2 tiles/step
      v8f O0, O1;
      #pragma unroll
      for (int r = 0; r < 8; ++r) { O0[r] = 0.0f; O1[r] = 0.0f; }
      #pragma unroll
      for (int kt = 0; kt < (MTILES + 1) / 2; ++kt) {
        #pragma unroll
        for (int tix = 0; tix < 2; ++tix) {
          const int nj = 2 * kt + tix;
          #pragma unroll
          for (int r = 0; r < 8; ++r)
            pb[(half8 + r) * 32 + tix * 16 + l15] =
                (nj < MTILES) ? (_Float16)S[nj][r] : (_Float16)0.0f;
        }
        asm volatile("s_wait_dscnt 0x0" ::: "memory");  // intra-wave LDS RAW
        v16h pf = load_a_frag(pb, 32, 0, 0);
        v16h v0 = load_b_frag(bufV, MTPS, h * DH, kt * 32);
        v16h v1 = load_b_frag(bufV, MTPS, h * DH + 16, kt * 32);
        O0 = __builtin_amdgcn_wmma_f32_16x16x32_f16(false, pf, false, v0, (short)0, O0, false, false);
        O1 = __builtin_amdgcn_wmma_f32_16x16x32_f16(false, pf, false, v1, (short)0, O1, false, false);
      }
      #pragma unroll
      for (int r = 0; r < 8; ++r) {
        const float inv = 1.0f / rsum[r];
        const int m = mi * 16 + half8 + r;
        bufB[m * D + h * DH + l15] = (_Float16)(O0[r] * inv);
        bufB[m * D + h * DH + 16 + l15] = (_Float16)(O1[r] * inv);
      }
    }
  }
  __syncthreads();

  // ---- Phase 5: out = O @ Wout^T + b_out, scatter to d_out rows ----
  for (int t = wv; t < MTILES * NT; t += NWAVES) {
    const int mi = t / NT, ni = t % NT;
    const float bias = b_out[ni * 16 + l15];
    v8f acc;
    #pragma unroll
    for (int r = 0; r < 8; ++r) acc[r] = bias;
    #pragma unroll
    for (int k = 0; k < D; k += 32) {
      v16h a = load_a_frag(bufB, D, mi * 16, k);
      v16h b = load_b_frag(Wout16, D, ni * 16, k);
      acc = __builtin_amdgcn_wmma_f32_16x16x32_f16(false, a, false, b, (short)0, acc, false, false);
    }
    #pragma unroll
    for (int r = 0; r < 8; ++r) {
      const int m = mi * 16 + half8 + r;
      if (m < MT) {
        const int row = s2r[win * MT + m];
        if (row >= 0) out[(long)row * D + ni * 16 + l15] = acc[r];
      }
    }
  }
}

// ---------------------------------------------------------------------------
extern "C" void kernel_launch(void* const* d_in, const int* in_sizes, int n_in,
                              void* d_out, int out_size, void* d_ws, size_t ws_size,
                              hipStream_t stream) {
  const float* feat  = (const float*)d_in[0];
  const float* pos_a = (const float*)d_in[1];
  const float* pos_b = (const float*)d_in[2];
  const float* W_in  = (const float*)d_in[3];
  const float* b_in  = (const float*)d_in[4];
  const float* W_out = (const float*)d_in[5];
  const float* b_out = (const float*)d_in[6];
  const int* idx_a  = (const int*)d_in[7];
  const int* idx_b  = (const int*)d_in[8];
  const int* slot_a = (const int*)d_in[9];
  const int* slot_b = (const int*)d_in[10];
  float* out = (float*)d_out;

  char* ws = (char*)d_ws;
  int* s2rA = (int*)ws;                               // 72000 ints
  int* s2rB = (int*)(ws + NW_A * MT_A * 4);           // 57600 ints
  const size_t w16_off = (size_t)(NW_A * MT_A + NW_B * MT_B) * 4;  // 518400 (16B aligned)
  _Float16* Win16  = (_Float16*)(ws + w16_off);                     // 768*256 halves
  _Float16* Wout16 = (_Float16*)(ws + w16_off + (size_t)3 * D * D * 2);

  const int s2r_tot = NW_A * MT_A + NW_B * MT_B;  // 129600
  wa_init_s2r<<<(s2r_tot + 255) / 256, 256, 0, stream>>>(s2rA, s2r_tot);
  wa_scatter<<<(NV_A + NV_B + 255) / 256, 256, 0, stream>>>(s2rA, s2rB, idx_a, idx_b, slot_a, slot_b);
  wa_cvt_weights<<<(3 * D * D + 255) / 256, 256, 0, stream>>>(W_in, W_out, Win16, Wout16);

  auto smem_bytes = [](int MTP, int MTPS) -> size_t {
    return ((size_t)(3 * MTP * D + D * MTPS + NWAVES * 512)) * 2 + (size_t)MTP * 4;
  };
  const size_t smA = smem_bytes(48, 64);     // ~115 KB
  const size_t smB = smem_bytes(144, 160);   // ~305 KB  (fits 320 KB/WGP LDS)

  (void)hipFuncSetAttribute((const void*)wa_attn<MT_A, 48, 64>,
                            hipFuncAttributeMaxDynamicSharedMemorySize, (int)smA);
  (void)hipFuncSetAttribute((const void*)wa_attn<MT_B, 144, 160>,
                            hipFuncAttributeMaxDynamicSharedMemorySize, (int)smB);

  wa_attn<MT_A, 48, 64><<<NW_A, 256, smA, stream>>>(
      feat, pos_a, b_in, b_out, s2rA, Win16, Wout16, out);
  wa_attn<MT_B, 144, 160><<<NW_B, 256, smB, stream>>>(
      feat, pos_b, b_in, b_out, s2rB, Win16, Wout16, out);
}
